// CaMoE_Block_40072044871825
// MI455X (gfx1250) — matrix operand order repaired and checked by reading, not compiled, not run on hardware
//
#include <hip/hip_runtime.h>
#include <cstdint>

// ---------------------------------------------------------------------------
// CDNA5 (gfx1250) CaMoE block: bf16 WMMA GEMM pipeline (double-buffered LDS,
// async global->LDS DMA) + RWKV scan + routing.
// wave32 everywhere; WMMA shape 16x16x32 bf16 -> f32 accumulate.
// ---------------------------------------------------------------------------

typedef __attribute__((ext_vector_type(16))) __bf16 v16bf;
typedef __attribute__((ext_vector_type(8)))  float  v8f;
typedef int v4i_ __attribute__((vector_size(16)));   // matches builtin proto

#define AS1 __attribute__((address_space(1)))
#define AS3 __attribute__((address_space(3)))

#if defined(__has_builtin)
#if __has_builtin(__builtin_amdgcn_global_load_async_to_lds_b128)
#define HAVE_ASYNC_LDS 1
#endif
#endif
#ifndef HAVE_ASYNC_LDS
#define HAVE_ASYNC_LDS 0
#endif

__device__ __forceinline__ void wait_async0() {
#if defined(__has_builtin) && __has_builtin(__builtin_amdgcn_s_wait_asynccnt)
  __builtin_amdgcn_s_wait_asynccnt(0);
#else
  asm volatile("s_wait_asynccnt 0" ::: "memory");
#endif
}

__device__ __forceinline__ __bf16 f2bf(float f) {
  unsigned u = __builtin_bit_cast(unsigned, f);
  unsigned r = (u + 0x7FFFu + ((u >> 16) & 1u)) >> 16;   // round-nearest-even
  unsigned short s = (unsigned short)r;
  return __builtin_bit_cast(__bf16, s);
}
__device__ __forceinline__ float bf2f(__bf16 b) {
  unsigned short s = __builtin_bit_cast(unsigned short, b);
  unsigned u = ((unsigned)s) << 16;
  return __builtin_bit_cast(float, u);
}

// ---------------------------------------------------------------------------
// Weight convert + transpose: fp32 W[K,N] -> bf16 Wt[N,K]
// ---------------------------------------------------------------------------
__global__ __launch_bounds__(256) void cvt_transpose_kernel(
    const float* __restrict__ W, __bf16* __restrict__ Wt, int K, int N) {
  const size_t i = (size_t)blockIdx.x * 256 + threadIdx.x;
  if (i >= (size_t)K * N) return;
  const int k = (int)(i % (size_t)K);
  const int n = (int)(i / (size_t)K);
  Wt[i] = f2bf(W[(size_t)k * N + n]);
}

// ---------------------------------------------------------------------------
// LayerNorm over C=1024, one block per row, bf16 output with stride ldo.
// ---------------------------------------------------------------------------
__global__ __launch_bounds__(256) void ln_kernel(
    const float* __restrict__ x, const float* __restrict__ g,
    const float* __restrict__ b, __bf16* __restrict__ out, int ldo) {
  const int m = blockIdx.x;
  const int tid = threadIdx.x;
  const float* xr = x + (size_t)m * 1024;
  float v[4], s = 0.f, sq = 0.f;
#pragma unroll
  for (int j = 0; j < 4; ++j) {
    v[j] = xr[tid + j * 256];
    s += v[j]; sq += v[j] * v[j];
  }
  __shared__ float rs[256], rq[256];
  rs[tid] = s; rq[tid] = sq;
  __syncthreads();
  for (int off = 128; off > 0; off >>= 1) {
    if (tid < off) { rs[tid] += rs[tid + off]; rq[tid] += rq[tid + off]; }
    __syncthreads();
  }
  const float mean = rs[0] * (1.f / 1024.f);
  const float var  = rq[0] * (1.f / 1024.f) - mean * mean;
  const float inv  = rsqrtf(var + 1e-5f);
#pragma unroll
  for (int j = 0; j < 4; ++j) {
    const int c = tid + j * 256;
    out[(size_t)m * ldo + c] = f2bf((v[j] - mean) * inv * g[c] + b[c]);
  }
}

// ---------------------------------------------------------------------------
// bf16 WMMA GEMM: C[M,N] = A[M,K] (row-major, stride lda) x B (given as
// Bt[N,K], i.e. B transposed so tiles are contiguous). Block tile 128x128,
// K-tile 32, double-buffered LDS; global->LDS via async DMA when available.
// 8 waves; each wave computes 32 rows x 64 cols = 2x4 WMMA tiles.
// Epilogues (EPI):
//   0: outF = acc
//   1: outF = acc + res            (residual add, fp32)
//   2: outB = bf16(relu(acc)^2)    (FFN expert hidden)
//   3: outF += gate[row] * acc     (gated MoE accumulate)
//   4: outB = bf16(acc)
//   5: outF = acc; outB = bf16(acc) (dual: v projection)
// ---------------------------------------------------------------------------
template <int EPI>
__global__ __launch_bounds__(256) void gemm_bf16_kernel(
    const __bf16* __restrict__ A, int lda,
    const __bf16* __restrict__ Bt,
    float* __restrict__ outF, __bf16* __restrict__ outB,
    const float* __restrict__ res, const float* __restrict__ gate,
    int M, int N, int K) {
  __shared__ __align__(16) __bf16 lA[2][128 * 32];
  __shared__ __align__(16) __bf16 lB[2][128 * 32];
  const int tid  = threadIdx.x;
  const int wave = tid >> 5;
  const int lane = tid & 31;
  const int row0 = blockIdx.x * 128;
  const int col0 = blockIdx.y * 128;
  const int wm = (wave & 3) * 32;    // wave's row offset in block tile
  const int wn = (wave >> 2) * 64;   // wave's col offset in block tile
  const int hk = lane >> 4;          // lane half (K-group select)
  const int rr = lane & 15;          // row/col within 16

  // This thread's two 16-byte chunks of each 128x32 tile (512 chunks total).
  const int r0 = tid >> 2,          cc0 = (tid & 3) * 8;
  const int r1 = (tid + 256) >> 2,  cc1 = ((tid + 256) & 3) * 8;

  v8f acc[2][4] = {};
  const int nK = K >> 5;

  // Issue the global->LDS copy of K-tile `kt` into LDS buffer `buf`.
  auto issue_tile = [&](int kt, int buf) {
    const int kb = kt * 32;
#if HAVE_ASYNC_LDS
    __builtin_amdgcn_global_load_async_to_lds_b128(
        (AS1 v4i_*)(A + (size_t)(row0 + r0) * lda + kb + cc0),
        (AS3 v4i_*)&lA[buf][r0 * 32 + cc0], 0, 0);
    __builtin_amdgcn_global_load_async_to_lds_b128(
        (AS1 v4i_*)(A + (size_t)(row0 + r1) * lda + kb + cc1),
        (AS3 v4i_*)&lA[buf][r1 * 32 + cc1], 0, 0);
    __builtin_amdgcn_global_load_async_to_lds_b128(
        (AS1 v4i_*)(Bt + (size_t)(col0 + r0) * K + kb + cc0),
        (AS3 v4i_*)&lB[buf][r0 * 32 + cc0], 0, 0);
    __builtin_amdgcn_global_load_async_to_lds_b128(
        (AS1 v4i_*)(Bt + (size_t)(col0 + r1) * K + kb + cc1),
        (AS3 v4i_*)&lB[buf][r1 * 32 + cc1], 0, 0);
#else
    *(uint4*)&lA[buf][r0 * 32 + cc0] =
        *(const uint4*)(A + (size_t)(row0 + r0) * lda + kb + cc0);
    *(uint4*)&lA[buf][r1 * 32 + cc1] =
        *(const uint4*)(A + (size_t)(row0 + r1) * lda + kb + cc1);
    *(uint4*)&lB[buf][r0 * 32 + cc0] =
        *(const uint4*)(Bt + (size_t)(col0 + r0) * K + kb + cc0);
    *(uint4*)&lB[buf][r1 * 32 + cc1] =
        *(const uint4*)(Bt + (size_t)(col0 + r1) * K + kb + cc1);
#endif
  };

  issue_tile(0, 0);   // prologue: tile 0 in flight into buffer 0

  for (int kt = 0; kt < nK; ++kt) {
    const int buf = kt & 1;
#if HAVE_ASYNC_LDS
    wait_async0();        // this wave's chunks of tile kt have landed
#endif
    // All waves' chunks landed; also: everyone is done reading buf from the
    // previous round, so refilling it (for tile kt+2, issued next iteration)
    // is safe. One barrier per K-step.
    __syncthreads();
    if (kt + 1 < nK) issue_tile(kt + 1, buf ^ 1);

    v16bf af[2], bfr[4];
#pragma unroll
    for (int i = 0; i < 2; ++i) {
      // A 16x32 bf16: lanes0-15 row M, K0..15; lanes16-31 K16..31
      const __bf16* p = &lA[buf][(wm + i * 16 + rr) * 32];
      union { uint4 u[2]; v16bf v; } t;
      t.u[0] = *(const uint4*)(p + hk * 16);
      t.u[1] = *(const uint4*)(p + hk * 16 + 8);
      af[i] = t.v;
    }
#pragma unroll
    for (int j = 0; j < 4; ++j) {
      // B 32x16 bf16: lane holds one column; lanes0-15 K0..15, lanes16-31 K16..31
      const __bf16* p = &lB[buf][(wn + j * 16 + rr) * 32 + hk * 16];
      union { uint4 u[2]; v16bf v; } t;
      t.u[0] = *(const uint4*)(p);
      t.u[1] = *(const uint4*)(p + 8);
      bfr[j] = t.v;
    }
#pragma unroll
    for (int i = 0; i < 2; ++i)
#pragma unroll
      for (int j = 0; j < 4; ++j)
        acc[i][j] = __builtin_amdgcn_wmma_f32_16x16x32_bf16(
            false, af[i], false, bfr[j], (short)0, acc[i][j], false, false);
  }

  // Epilogue. C/D layout: VGPR r -> (M = r + 8*hk, N = rr) within a 16x16 tile.
#pragma unroll
  for (int i = 0; i < 2; ++i)
#pragma unroll
    for (int j = 0; j < 4; ++j)
#pragma unroll
      for (int rg = 0; rg < 8; ++rg) {
        const int row = row0 + wm + i * 16 + hk * 8 + rg;
        const int col = col0 + wn + j * 16 + rr;
        const size_t off = (size_t)row * N + col;
        const float v = acc[i][j][rg];
        if constexpr (EPI == 0) {
          outF[off] = v;
        } else if constexpr (EPI == 1) {
          outF[off] = v + res[off];
        } else if constexpr (EPI == 2) {
          const float t = v > 0.f ? v : 0.f;
          outB[off] = f2bf(t * t);
        } else if constexpr (EPI == 3) {
          outF[off] += gate[row] * v;
        } else if constexpr (EPI == 4) {
          outB[off] = f2bf(v);
        } else {
          outF[off] = v;
          outB[off] = f2bf(v);
        }
      }
}

// ---------------------------------------------------------------------------
// RWKV7-style linear attention scan. One block per (batch, head).
// State S[d][e] (64x64 fp32): thread owns column e = tid&63 and 16 d's.
//   S = S * sigmoid(w)[d] + k[d]*v[e];  out[e] = sum_d r[d]*S[d][e]
// out written as bf16 into hcat second half (stride 2048).
// ---------------------------------------------------------------------------
__global__ __launch_bounds__(256) void rwkv_scan_kernel(
    const __bf16* __restrict__ kb, const __bf16* __restrict__ vb,
    const __bf16* __restrict__ rb, const float* __restrict__ w_att,
    __bf16* __restrict__ outB) {
  const int bh = blockIdx.x;
  const int b = bh >> 4;
  const int h = bh & 15;
  const int tid = threadIdx.x;
  const int e0 = tid & 63;
  const int db = (tid >> 6) * 16;
  float dec[16], s[16];
#pragma unroll
  for (int j = 0; j < 16; ++j) {
    dec[j] = 1.f / (1.f + expf(-w_att[h * 64 + db + j]));
    s[j] = 0.f;
  }
  __shared__ float sk[64], sv[64], sr[64], sout[64];
  if (tid < 64) sout[tid] = 0.f;
  __syncthreads();
  for (int t = 0; t < 1024; ++t) {
    const size_t base = ((size_t)(b * 1024 + t)) * 1024 + h * 64;
    if (tid < 64)       sk[tid]       = bf2f(kb[base + tid]);
    else if (tid < 128) sv[tid - 64]  = bf2f(vb[base + tid - 64]);
    else if (tid < 192) sr[tid - 128] = bf2f(rb[base + tid - 128]);
    __syncthreads();
    const float vv = sv[e0];
    float po = 0.f;
#pragma unroll
    for (int j = 0; j < 16; ++j) {
      s[j] = s[j] * dec[j] + sk[db + j] * vv;
      po += sr[db + j] * s[j];
    }
    atomicAdd(&sout[e0], po);   // ds_add_f32 across the 4 d-groups
    __syncthreads();
    if (tid < 64) {
      outB[((size_t)(b * 1024 + t)) * 2048 + h * 64 + tid] = f2bf(sout[tid]);
      sout[tid] = 0.f;          // same thread reads then resets: race-free
    }
    __syncthreads();
  }
}

// ---------------------------------------------------------------------------
// Routing: per token, 17 dot products of length 1024 (8 conf, 1 difficulty,
// 8 affinity), then bids + top-2 softmax -> dense gate array gates[E][M].
// ---------------------------------------------------------------------------
__global__ __launch_bounds__(256) void routing_kernel(
    const __bf16* __restrict__ hcat, const float* __restrict__ Wc_rwkv,
    const float* __restrict__ Wc_trans, const float* __restrict__ Wd,
    const float* __restrict__ Wa, const float* __restrict__ cap,
    float* __restrict__ gates, int M) {
  const int m = blockIdx.x;
  const int tid = threadIdx.x;
  const int wave = tid >> 5, lane = tid & 31;
  const __bf16* h = hcat + (size_t)m * 2048;
  float part[17];
#pragma unroll
  for (int i = 0; i < 17; ++i) part[i] = 0.f;
  for (int j = 0; j < 4; ++j) {
    const int c = tid + j * 256;
    const float hv = bf2f(h[c]);
#pragma unroll
    for (int e = 0; e < 6; ++e) part[e] += hv * Wc_rwkv[e * 1024 + c];
    part[6] += hv * Wc_trans[c];
    part[7] += hv * Wc_trans[1024 + c];
    part[8] += hv * Wd[c];
#pragma unroll
    for (int e = 0; e < 8; ++e) part[9 + e] += hv * Wa[c * 8 + e];
  }
#pragma unroll
  for (int i = 0; i < 17; ++i)
    for (int off = 16; off > 0; off >>= 1)
      part[i] += __shfl_xor(part[i], off);
  __shared__ float red[8][17];
  if (lane == 0)
#pragma unroll
    for (int i = 0; i < 17; ++i) red[wave][i] = part[i];
  __syncthreads();
  if (tid == 0) {
    float tot[17];
#pragma unroll
    for (int i = 0; i < 17; ++i) {
      float s = 0.f;
      for (int w = 0; w < 8; ++w) s += red[w][i];
      tot[i] = s;
    }
    float conf[8];
#pragma unroll
    for (int e = 0; e < 8; ++e) conf[e] = 1.f / (1.f + expf(-tot[e]));
    const float diff = log1pf(expf(tot[8]));            // softplus
    float mx = tot[9];
    for (int e = 1; e < 8; ++e) mx = fmaxf(mx, tot[9 + e]);
    float sm[8], ssum = 0.f;
#pragma unroll
    for (int e = 0; e < 8; ++e) { sm[e] = expf(tot[9 + e] - mx); ssum += sm[e]; }
    float bids[8];
#pragma unroll
    for (int e = 0; e < 8; ++e) bids[e] = conf[e] * cap[e] * diff + sm[e] / ssum;
    int i1 = 0;
    for (int e = 1; e < 8; ++e) if (bids[e] > bids[i1]) i1 = e;
    int i2 = (i1 == 0) ? 1 : 0;
    for (int e = 0; e < 8; ++e) if (e != i1 && bids[e] > bids[i2]) i2 = e;
    const float e2 = expf(bids[i2] - bids[i1]);
    const float w1 = 1.f / (1.f + e2);
#pragma unroll
    for (int e = 0; e < 8; ++e) gates[(size_t)e * M + m] = 0.f;
    gates[(size_t)i1 * M + m] = w1;
    gates[(size_t)i2 * M + m] = e2 * w1;
  }
}

// ---------------------------------------------------------------------------
// Transformer-expert cross-attention over P=4 bridge-prefix slots.
// One block per token; 16 threads per head, each owning 4 of the 64 d's.
// ---------------------------------------------------------------------------
__global__ __launch_bounds__(256) void prefix_attn_kernel(
    const __bf16* __restrict__ q, const __bf16* __restrict__ kp,
    const __bf16* __restrict__ vp, __bf16* __restrict__ o) {
  const int m = blockIdx.x;
  const int t = threadIdx.x;
  const int h = t >> 4;
  const int dl = t & 15;
  const size_t qbase = (size_t)m * 1024 + h * 64;
  const size_t pbase = (size_t)m * 4096 + h * 64;   // kp/vp: [m*4+p, 1024]
  float qv[4];
#pragma unroll
  for (int jj = 0; jj < 4; ++jj) qv[jj] = bf2f(q[qbase + dl + jj * 16]);
  float dot[4];
#pragma unroll
  for (int p = 0; p < 4; ++p) {
    float a = 0.f;
#pragma unroll
    for (int jj = 0; jj < 4; ++jj)
      a += qv[jj] * bf2f(kp[pbase + (size_t)p * 1024 + dl + jj * 16]);
    dot[p] = a;
  }
#pragma unroll
  for (int p = 0; p < 4; ++p) {
    dot[p] += __shfl_xor(dot[p], 1);
    dot[p] += __shfl_xor(dot[p], 2);
    dot[p] += __shfl_xor(dot[p], 4);
    dot[p] += __shfl_xor(dot[p], 8);
    dot[p] *= 0.125f;                 // 1/sqrt(64)
  }
  const float mx = fmaxf(fmaxf(dot[0], dot[1]), fmaxf(dot[2], dot[3]));
  float att[4], sum = 0.f;
#pragma unroll
  for (int p = 0; p < 4; ++p) { att[p] = expf(dot[p] - mx); sum += att[p]; }
  const float inv = 1.f / sum;
#pragma unroll
  for (int jj = 0; jj < 4; ++jj) {
    float ov = 0.f;
#pragma unroll
    for (int p = 0; p < 4; ++p)
      ov += att[p] * bf2f(vp[pbase + (size_t)p * 1024 + dl + jj * 16]);
    o[qbase + dl + jj * 16] = f2bf(ov * inv);
  }
}

// ---------------------------------------------------------------------------
// Launch pipeline.
// ---------------------------------------------------------------------------
extern "C" void kernel_launch(void* const* d_in, const int* in_sizes, int n_in,
                              void* d_out, int out_size, void* d_ws, size_t ws_size,
                              hipStream_t stream) {
  (void)in_sizes; (void)n_in; (void)out_size; (void)ws_size;
  const int M = 8 * 1024;   // B*T
  const int C = 1024, FF = 4096;

  const float* x        = (const float*)d_in[0];
  const float* cap      = (const float*)d_in[2];
  const float* g1       = (const float*)d_in[3];
  const float* b1       = (const float*)d_in[4];
  const float* g2       = (const float*)d_in[5];
  const float* b2       = (const float*)d_in[6];
  const float* Wr       = (const float*)d_in[7];
  const float* Wk       = (const float*)d_in[8];
  const float* Wv       = (const float*)d_in[9];
  const float* Wo       = (const float*)d_in[10];
  const float* w_att    = (const float*)d_in[11];
  const float* Wb       = (const float*)d_in[12];
  const float* Wk_ffn   = (const float*)d_in[13];
  const float* Wv_ffn   = (const float*)d_in[14];
  const float* Wc_rwkv  = (const float*)d_in[15];
  const float* Wq_t     = (const float*)d_in[16];
  const float* Wk_t     = (const float*)d_in[17];
  const float* Wv_t     = (const float*)d_in[18];
  const float* Wo_t     = (const float*)d_in[19];
  const float* Wc_trans = (const float*)d_in[20];
  const float* Wd       = (const float*)d_in[21];
  const float* Wa       = (const float*)d_in[22];

  float* out0 = (float*)d_out;                   // x_after_att + final_out
  float* out1 = (float*)d_out + (size_t)M * C;   // v_first_out

  char* wp = (char*)d_ws;
  auto alloc = [&](size_t bytes) -> void* {
    void* r = (void*)wp;
    wp += (bytes + 255) & ~(size_t)255;
    return r;
  };
  __bf16* xln  = (__bf16*)alloc((size_t)M * C * 2);
  __bf16* hcat = (__bf16*)alloc((size_t)M * 2 * C * 2);  // [h | rwkv_state] bf16
  __bf16* rB   = (__bf16*)alloc((size_t)M * C * 2);
  __bf16* kB   = (__bf16*)alloc((size_t)M * C * 2);
  __bf16* vB   = (__bf16*)alloc((size_t)M * C * 2);
  __bf16* pref = (__bf16*)alloc((size_t)M * FF * 2);     // [M,4,C] bf16
  __bf16* kk   = (__bf16*)alloc((size_t)M * FF * 2);
  __bf16* qB   = (__bf16*)alloc((size_t)M * C * 2);
  __bf16* kpB  = (__bf16*)alloc((size_t)M * FF * 2);
  __bf16* vpB  = (__bf16*)alloc((size_t)M * FF * 2);
  __bf16* oB   = (__bf16*)alloc((size_t)M * C * 2);
  float*  gates = (float*)alloc((size_t)8 * M * 4);
  __bf16* WrT  = (__bf16*)alloc((size_t)C * C * 2);
  __bf16* WkT  = (__bf16*)alloc((size_t)C * C * 2);
  __bf16* WvT  = (__bf16*)alloc((size_t)C * C * 2);
  __bf16* WoT  = (__bf16*)alloc((size_t)C * C * 2);
  __bf16* WbT  = (__bf16*)alloc((size_t)FF * 2 * C * 2);
  __bf16* WkfT = (__bf16*)alloc((size_t)6 * FF * C * 2);
  __bf16* WvfT = (__bf16*)alloc((size_t)6 * C * FF * 2);
  __bf16* WqT  = (__bf16*)alloc((size_t)2 * C * C * 2);
  __bf16* WktT = (__bf16*)alloc((size_t)2 * C * C * 2);
  __bf16* WvtT = (__bf16*)alloc((size_t)2 * C * C * 2);
  __bf16* WotT = (__bf16*)alloc((size_t)2 * C * C * 2);

  auto cvt = [&](const float* W, __bf16* Wt, int K, int N) {
    const size_t total = (size_t)K * N;
    cvt_transpose_kernel<<<dim3((unsigned)((total + 255) / 256)), dim3(256), 0,
                           stream>>>(W, Wt, K, N);
  };
  cvt(Wr, WrT, C, C); cvt(Wk, WkT, C, C); cvt(Wv, WvT, C, C); cvt(Wo, WoT, C, C);
  cvt(Wb, WbT, 2 * C, FF);
  for (int e = 0; e < 6; ++e) {
    cvt(Wk_ffn + (size_t)e * C * FF, WkfT + (size_t)e * FF * C, C, FF);
    cvt(Wv_ffn + (size_t)e * FF * C, WvfT + (size_t)e * C * FF, FF, C);
  }
  for (int e = 0; e < 2; ++e) {
    cvt(Wq_t + (size_t)e * C * C, WqT  + (size_t)e * C * C, C, C);
    cvt(Wk_t + (size_t)e * C * C, WktT + (size_t)e * C * C, C, C);
    cvt(Wv_t + (size_t)e * C * C, WvtT + (size_t)e * C * C, C, C);
    cvt(Wo_t + (size_t)e * C * C, WotT + (size_t)e * C * C, C, C);
  }

  auto gg = [](int m, int n) {
    return dim3((unsigned)(m / 128), (unsigned)(n / 128));
  };

  // ---- attention stage ----
  ln_kernel<<<dim3(M), dim3(256), 0, stream>>>(x, g1, b1, xln, C);
  gemm_bf16_kernel<4><<<gg(M, C), 256, 0, stream>>>(xln, C, WrT, nullptr, rB,
                                                    nullptr, nullptr, M, C, C);
  gemm_bf16_kernel<4><<<gg(M, C), 256, 0, stream>>>(xln, C, WkT, nullptr, kB,
                                                    nullptr, nullptr, M, C, C);
  gemm_bf16_kernel<5><<<gg(M, C), 256, 0, stream>>>(xln, C, WvT, out1, vB,
                                                    nullptr, nullptr, M, C, C);
  rwkv_scan_kernel<<<dim3(128), dim3(256), 0, stream>>>(kB, vB, rB, w_att,
                                                        hcat + C);
  // att_out = rwkv_state @ Wo ; x_after_att = x + att_out -> d_out first half
  gemm_bf16_kernel<1><<<gg(M, C), 256, 0, stream>>>(hcat + C, 2 * C, WoT, out0,
                                                    nullptr, x, nullptr, M, C, C);
  ln_kernel<<<dim3(M), dim3(256), 0, stream>>>(out0, g2, b2, hcat, 2 * C);

  // ---- routing stage ----
  routing_kernel<<<dim3(M), dim3(256), 0, stream>>>(hcat, Wc_rwkv, Wc_trans, Wd,
                                                    Wa, cap, gates, M);

  // ---- expert stage ----
  gemm_bf16_kernel<4><<<gg(M, FF), 256, 0, stream>>>(hcat, 2 * C, WbT, nullptr,
                                                     pref, nullptr, nullptr,
                                                     M, FF, 2 * C);
  for (int e = 0; e < 6; ++e) {  // SparseRWKVFFN experts
    gemm_bf16_kernel<2><<<gg(M, FF), 256, 0, stream>>>(
        hcat, 2 * C, WkfT + (size_t)e * FF * C, nullptr, kk, nullptr, nullptr,
        M, FF, C);
    gemm_bf16_kernel<3><<<gg(M, C), 256, 0, stream>>>(
        kk, FF, WvfT + (size_t)e * C * FF, out0, nullptr, nullptr,
        gates + (size_t)e * M, M, C, FF);
  }
  for (int e = 0; e < 2; ++e) {  // LinearTransformer experts
    gemm_bf16_kernel<4><<<gg(M, C), 256, 0, stream>>>(
        hcat, 2 * C, WqT + (size_t)e * C * C, nullptr, qB, nullptr, nullptr,
        M, C, C);
    gemm_bf16_kernel<4><<<gg(4 * M, C), 256, 0, stream>>>(
        pref, C, WktT + (size_t)e * C * C, nullptr, kpB, nullptr, nullptr,
        4 * M, C, C);
    gemm_bf16_kernel<4><<<gg(4 * M, C), 256, 0, stream>>>(
        pref, C, WvtT + (size_t)e * C * C, nullptr, vpB, nullptr, nullptr,
        4 * M, C, C);
    prefix_attn_kernel<<<dim3(M), dim3(256), 0, stream>>>(qB, kpB, vpB, oB);
    gemm_bf16_kernel<3><<<gg(M, C), 256, 0, stream>>>(
        oB, C, WotT + (size_t)e * C * C, out0, nullptr, nullptr,
        gates + (size_t)(6 + e) * M, M, C, C);
  }
}
